// PuEVA_84164179132498
// MI455X (gfx1250) — compile-verified, matrix operand order checked by reading, not run on hardware
//
#include <hip/hip_runtime.h>
#include <hip/hip_bf16.h>
#include <math.h>

// ---------------------------------------------------------------------------
// PU-EVA forward for MI455X (gfx1250, wave32, WMMA).
// Heavy GEMMs (g/h/l 512x1027, r1/r2 515x515, r3 128x515) run on
// v_wmma_f32_16x16x32_f16 with f16 operands / f32 accumulation.
// Activations for GEMMs live in a K-packed f16 layout [K/32, P, 32] so every
// WMMA fragment load is a 16B-aligned global_load_b128 with no bounds checks
// (K and M are zero-padded at pack/build time).
// Small FEC convs (12-84 ch) + kNN + softmax stay on VALU (memory bound).
// ---------------------------------------------------------------------------

typedef __attribute__((ext_vector_type(16))) _Float16 v16h;
typedef __attribute__((ext_vector_type(8)))  _Float16 v8h;
typedef __attribute__((ext_vector_type(8)))  float    v8f;

#define EPSF 1e-5f

// packed index: channel c, position p, position-count P (halves)
__device__ __forceinline__ size_t pkidx(int c, size_t p, size_t P) {
    return (((size_t)(c >> 5)) * P + p) * 32 + (size_t)(c & 31);
}

// ---------------------------------------------------------------------------
// kNN: one block per (b,n); d2 to all N points in LDS, 17 argmin extractions
// (first hit is self, dropped). Tie-break = lowest index (matches stable topk).
// ---------------------------------------------------------------------------
__global__ void knn_kernel(const float* __restrict__ x, int* __restrict__ idx,
                           int N, int K, int B) {
    __shared__ float d2[1024];
    __shared__ float rv[256];
    __shared__ int   ri[256];
    int bn = blockIdx.x;
    int n = bn % N, b = bn / N;
    int tid = threadIdx.x;
    float qx = x[((size_t)b * 3 + 0) * N + n];
    float qy = x[((size_t)b * 3 + 1) * N + n];
    float qz = x[((size_t)b * 3 + 2) * N + n];
    for (int j = tid; j < N; j += 256) {
        float dx = x[((size_t)b * 3 + 0) * N + j] - qx;
        float dy = x[((size_t)b * 3 + 1) * N + j] - qy;
        float dz = x[((size_t)b * 3 + 2) * N + j] - qz;
        d2[j] = dx * dx + dy * dy + dz * dz;
    }
    __syncthreads();
    for (int s = 0; s < K + 1; ++s) {
        float bv = 3.4e38f; int bi = N;
        for (int j = tid; j < N; j += 256) {
            float v = d2[j];
            if (v < bv || (v == bv && j < bi)) { bv = v; bi = j; }
        }
        rv[tid] = bv; ri[tid] = bi;
        __syncthreads();
        for (int st = 128; st > 0; st >>= 1) {
            if (tid < st) {
                if (rv[tid + st] < rv[tid] ||
                    (rv[tid + st] == rv[tid] && ri[tid + st] < ri[tid])) {
                    rv[tid] = rv[tid + st]; ri[tid] = ri[tid + st];
                }
            }
            __syncthreads();
        }
        int win = ri[0];
        if (tid == 0) {
            if (s > 0) idx[((size_t)b * N + n) * K + (s - 1)] = win;
            d2[win] = 3.4e38f;
        }
        __syncthreads();
    }
}

// ---------------------------------------------------------------------------
// pre conv: out[b,o,n] = sum_c W[o,c]*f[b,c,n] + bias[o]
// ---------------------------------------------------------------------------
__global__ void conv_pre_kernel(const float* __restrict__ f, const float* __restrict__ W,
                                const float* __restrict__ bias, float* __restrict__ out,
                                int Cin, int Cout, int N, int B) {
    int i = blockIdx.x * blockDim.x + threadIdx.x;
    int total = B * Cout * N;
    if (i >= total) return;
    int n = i % N, o = (i / N) % Cout, b = i / (N * Cout);
    const float* fb = f + (size_t)b * Cin * N + n;
    const float* w  = W + (size_t)o * Cin;
    float s = bias ? bias[o] : 0.f;
    for (int c = 0; c < Cin; ++c) s += w[c] * fb[(size_t)c * N];
    out[i] = s;
}

// training-mode batchnorm statistics over (B, Len) per channel
__global__ void bn_stats_kernel(const float* __restrict__ xbuf, float* __restrict__ mean,
                                float* __restrict__ rstd, int C, int Len, int B) {
    __shared__ float ss[256], sq[256];
    int c = blockIdx.x, tid = threadIdx.x;
    float s = 0.f, q = 0.f;
    for (int b = 0; b < B; ++b) {
        const float* p = xbuf + ((size_t)b * C + c) * Len;
        for (int i = tid; i < Len; i += 256) { float v = p[i]; s += v; q += v * v; }
    }
    ss[tid] = s; sq[tid] = q;
    __syncthreads();
    for (int st = 128; st > 0; st >>= 1) {
        if (tid < st) { ss[tid] += ss[tid + st]; sq[tid] += sq[tid + st]; }
        __syncthreads();
    }
    if (tid == 0) {
        float cnt = (float)(B * Len);
        float m = ss[0] / cnt;
        float v = sq[0] / cnt - m * m;
        mean[c] = m;
        rstd[c] = rsqrtf(fmaxf(v, 0.f) + EPSF);
    }
}

__global__ void bn_apply_relu_kernel(float* __restrict__ xbuf, const float* __restrict__ mean,
                                     const float* __restrict__ rstd, const float* __restrict__ g,
                                     const float* __restrict__ bet, int C, int Len, int B) {
    int i = blockIdx.x * blockDim.x + threadIdx.x;
    int total = B * C * Len;
    if (i >= total) return;
    int c = (i / Len) % C;
    float v = (xbuf[i] - mean[c]) * rstd[c] * g[c] + bet[c];
    xbuf[i] = v > 0.f ? v : 0.f;
}

// y48[b,c,n,k]: c<24 -> broadcast pre; else gathered pre via idx
__global__ void group48_kernel(const float* __restrict__ pre, const int* __restrict__ idx,
                               float* __restrict__ y48, int N, int K, int B) {
    int i = blockIdx.x * blockDim.x + threadIdx.x;
    int total = B * 48 * N * K;
    if (i >= total) return;
    int k = i % K, n = (i / K) % N, c = (i / (K * N)) % 48, b = i / (K * N * 48);
    float v;
    if (c < 24) v = pre[((size_t)b * 24 + c) * N + n];
    else {
        int j = idx[((size_t)b * N + n) * K + k];
        v = pre[((size_t)b * 24 + (c - 24)) * N + j];
    }
    y48[i] = v;
}

// dense-block conv over concatenated sources [s0(c0) | s1(c1) | s2(c2)]
__global__ void conv_nbr_kernel(const float* __restrict__ s0, int c0,
                                const float* __restrict__ s1, int c1,
                                const float* __restrict__ s2, int c2,
                                const float* __restrict__ W, float* __restrict__ out,
                                int Cout, int NK, int B) {
    int i = blockIdx.x * blockDim.x + threadIdx.x;
    int total = B * Cout * NK;
    if (i >= total) return;
    int p = i % NK, o = (i / NK) % Cout, b = i / (NK * Cout);
    const float* w = W + (size_t)o * (c0 + c1 + c2);
    float s = 0.f;
    for (int c = 0; c < c0; ++c) s += w[c] * s0[((size_t)b * c0 + c) * NK + p];
    for (int c = 0; c < c1; ++c) s += w[c0 + c] * s1[((size_t)b * c1 + c) * NK + p];
    for (int c = 0; c < c2; ++c) s += w[c0 + c1 + c] * s2[((size_t)b * c2 + c) * NK + p];
    out[i] = s;
}

// maxpool over K of [m3|m2|m1|y48] then concat previous features
__global__ void pool_concat_kernel(const float* __restrict__ m3o, const float* __restrict__ m2o,
                                   const float* __restrict__ m1o, const float* __restrict__ y48,
                                   const float* __restrict__ fprev, int cprev,
                                   float* __restrict__ fout, int N, int K, int B) {
    int cout = 84 + cprev;
    int i = blockIdx.x * blockDim.x + threadIdx.x;
    int total = B * cout * N;
    if (i >= total) return;
    int n = i % N, c = (i / N) % cout, b = i / (N * cout);
    float v;
    if (c < 84) {
        const float* src; int cc, cs;
        if (c < 12)      { src = m3o; cc = c;      cs = 12; }
        else if (c < 24) { src = m2o; cc = c - 12; cs = 12; }
        else if (c < 36) { src = m1o; cc = c - 24; cs = 12; }
        else             { src = y48; cc = c - 36; cs = 48; }
        const float* p = src + (((size_t)b * cs + cc) * N + n) * K;
        v = p[0];
        for (int k = 1; k < K; ++k) v = fmaxf(v, p[k]);
    } else {
        v = fprev[((size_t)b * cprev + (c - 84)) * N + n];
    }
    fout[i] = v;
}

// ---------------------------------------------------------------------------
// Build the full 1027-channel F tensor directly in K-packed f16
// ([1056/32, N*K, 32] per batch, channels 1027..1055 zero). Channels:
// [f bcast(339) | f gather(339) | gather-bcast(339) | diff(3) | nbr(3) | x(3) | dist(1)]
// ---------------------------------------------------------------------------
__global__ void buildF_kernel(const float* __restrict__ x, const float* __restrict__ f,
                              const int* __restrict__ idx, _Float16* __restrict__ Fh,
                              int N, int K, int B) {
    int i = blockIdx.x * blockDim.x + threadIdx.x;
    int total = B * N * K;
    if (i >= total) return;
    int k = i % K, n = (i / K) % N, b = i / (N * K);
    const int FD = 339, CP = 1056;
    int j = idx[((size_t)b * N + n) * K + k];
    size_t P = (size_t)N * K;
    size_t p = (size_t)n * K + k;
    _Float16* Fb = Fh + (size_t)b * CP * P;
    for (int c = 0; c < FD; ++c) {
        float a = f[((size_t)b * FD + c) * N + n];
        float g = f[((size_t)b * FD + c) * N + j];
        Fb[pkidx(c, p, P)]          = (_Float16)a;
        Fb[pkidx(FD + c, p, P)]     = (_Float16)g;
        Fb[pkidx(2 * FD + c, p, P)] = (_Float16)(g - a);
    }
    float dd = 0.f;
    for (int d = 0; d < 3; ++d) {
        float xn = x[((size_t)b * 3 + d) * N + n];
        float xj = x[((size_t)b * 3 + d) * N + j];
        float df = xj - xn;
        dd += df * df;
        Fb[pkidx(1017 + d, p, P)] = (_Float16)df;
        Fb[pkidx(1020 + d, p, P)] = (_Float16)xj;
        Fb[pkidx(1023 + d, p, P)] = (_Float16)xn;
    }
    Fb[pkidx(1026, p, P)] = (_Float16)sqrtf(dd);
    for (int c = 1027; c < CP; ++c) Fb[pkidx(c, p, P)] = (_Float16)0.f;
}

// deterministic counter-hash RNG; per-(b,c,n) pick R smallest of K uniforms
// (stable argsort-take-R equivalent) and gather F -> r (both K-packed).
__device__ __forceinline__ unsigned wanghash(unsigned v) {
    v ^= v >> 16; v *= 0x7feb352dU; v ^= v >> 15; v *= 0x846ca68bU; v ^= v >> 16;
    return v;
}

__global__ void subsample_kernel(const _Float16* __restrict__ Fh, _Float16* __restrict__ rh,
                                 int N, int K, int R, int B) {
    const int CF = 1027, CP = 1056;
    int i = blockIdx.x * blockDim.x + threadIdx.x;
    int total = B * CP * N;
    if (i >= total) return;
    int n = i % N, c = (i / N) % CP, b = i / (N * CP);
    size_t PF = (size_t)N * K, PR = (size_t)N * R;
    _Float16* rb = rh + (size_t)b * CP * PR;
    if (c >= CF) {  // zero pad channels
        for (int j = 0; j < R; ++j) rb[pkidx(c, (size_t)n * R + j, PR)] = (_Float16)0.f;
        return;
    }
    const _Float16* Fb = Fh + (size_t)b * CP * PF;
    unsigned seed = (unsigned)(((b * CF + c) * N + n) * 16);
    float u[16];
    bool used[16];
    for (int k = 0; k < K; ++k) {
        unsigned h = wanghash(seed + (unsigned)k + 0x9e3779b9u);
        u[k] = (float)(h >> 8) * (1.f / 16777216.f);
        used[k] = false;
    }
    for (int j = 0; j < R; ++j) {
        float best = 3.4e38f; int bi = 0;
        for (int k = 0; k < K; ++k)
            if (!used[k] && u[k] < best) { best = u[k]; bi = k; }
        used[bi] = true;
        rb[pkidx(c, (size_t)n * R + j, PR)] = Fb[pkidx(c, (size_t)n * K + bi, PF)];
    }
}

// ---------------------------------------------------------------------------
// WMMA GEMM.  Out(f16, K-packed [Mout/32,P,32]) =
//   opt_relu( W(f16,[Mp,Kp] zero-padded) @ X(f16, K-packed [Kp/32,P,32]) + bias )
// One wave -> 16(M) x 64(P) strip, 4 tiles share the A fragment.
// All K/M padding is zero-filled so the inner loop has NO conditionals:
// per k-step = 2x b128 A loads + 8x b128 B loads + 4x WMMA.
// A packing follows the documented 16-bit 16x32 layout (lanes 0-15 hold
// K {0..7,16..23}, lanes 16-31 hold K {8..15,24..31}); B mirrors the same
// per-lane K ordering by column.
// ---------------------------------------------------------------------------
__global__ void __launch_bounds__(256)
wmma_gemm_kernel(const _Float16* __restrict__ Wh,
                 const _Float16* __restrict__ Xh,
                 const float* __restrict__ bias,
                 _Float16* __restrict__ Out,
                 int M, int Kp, int P, int Mout, int relu) {
    int bz = blockIdx.z;
    const _Float16* X = Xh + (size_t)bz * Kp * P;
    _Float16* O = Out + (size_t)bz * Mout * P;

    int wave = threadIdx.x >> 5;   // uniform within wave
    int lane = threadIdx.x & 31;
    int m0 = (blockIdx.y * 8 + wave) * 16;
    if (m0 >= Mout) return;        // wave-uniform exit; EXEC all-1 around WMMA
    int p0 = blockIdx.x * 64;

    int lm  = lane & 15;
    int sel = lane >> 4;
    int row = m0 + lm;             // always < Mp (weights padded to tile grid)

    const v8h* ap = (const v8h*)(Wh + (size_t)row * Kp + sel * 8);
    size_t xoff = ((size_t)(p0 + lm)) * 32 + sel * 8;
    const int KB = Kp >> 5;

    v8f acc0 = {}, acc1 = {}, acc2 = {}, acc3 = {};

    for (int kb = 0; kb < KB; ++kb) {
        // A fragment: two contiguous 8-half runs of row `row`
        v8h alo = ap[0], ahi = ap[2];
        v16h a = __builtin_shufflevector(alo, ahi, 0, 1, 2, 3, 4, 5, 6, 7,
                                         8, 9, 10, 11, 12, 13, 14, 15);
        // B fragments: packed layout -> two b128 per tile
        const v8h* xp0 = (const v8h*)(X + (size_t)kb * P * 32 + xoff);
        const v8h* xp1 = (const v8h*)(X + (size_t)kb * P * 32 + xoff + 16 * 32);
        const v8h* xp2 = (const v8h*)(X + (size_t)kb * P * 32 + xoff + 32 * 32);
        const v8h* xp3 = (const v8h*)(X + (size_t)kb * P * 32 + xoff + 48 * 32);
        __builtin_prefetch(X + (size_t)(kb + 1) * P * 32 + xoff, 0, 1);
        v16h b0 = __builtin_shufflevector(xp0[0], xp0[2], 0, 1, 2, 3, 4, 5, 6, 7,
                                          8, 9, 10, 11, 12, 13, 14, 15);
        v16h b1 = __builtin_shufflevector(xp1[0], xp1[2], 0, 1, 2, 3, 4, 5, 6, 7,
                                          8, 9, 10, 11, 12, 13, 14, 15);
        v16h b2 = __builtin_shufflevector(xp2[0], xp2[2], 0, 1, 2, 3, 4, 5, 6, 7,
                                          8, 9, 10, 11, 12, 13, 14, 15);
        v16h b3 = __builtin_shufflevector(xp3[0], xp3[2], 0, 1, 2, 3, 4, 5, 6, 7,
                                          8, 9, 10, 11, 12, 13, 14, 15);
        acc0 = __builtin_amdgcn_wmma_f32_16x16x32_f16(false, a, false, b0, (short)0, acc0, false, false);
        acc1 = __builtin_amdgcn_wmma_f32_16x16x32_f16(false, a, false, b1, (short)0, acc1, false, false);
        acc2 = __builtin_amdgcn_wmma_f32_16x16x32_f16(false, a, false, b2, (short)0, acc2, false, false);
        acc3 = __builtin_amdgcn_wmma_f32_16x16x32_f16(false, a, false, b3, (short)0, acc3, false, false);
        ap += 4;                   // advance 32 halves
    }

    // D layout: lane l, component j -> m = m0 + j + 8*(l>>4), n = p0 + t*16 + lm
#pragma unroll
    for (int j = 0; j < 8; ++j) {
        int m = m0 + j + 8 * sel;
        if (m < Mout) {
            float bv = (bias && m < M) ? bias[m] : 0.f;
            float v0 = acc0[j] + bv, v1 = acc1[j] + bv, v2 = acc2[j] + bv, v3 = acc3[j] + bv;
            if (relu) {
                v0 = v0 > 0.f ? v0 : 0.f;
                v1 = v1 > 0.f ? v1 : 0.f;
                v2 = v2 > 0.f ? v2 : 0.f;
                v3 = v3 > 0.f ? v3 : 0.f;
            }
            size_t oa = ((size_t)(m >> 5) * P + (p0 + lm)) * 32 + (m & 31);
            O[oa]            = (_Float16)v0;
            O[oa + 16 * 32]  = (_Float16)v1;
            O[oa + 32 * 32]  = (_Float16)v2;
            O[oa + 48 * 32]  = (_Float16)v3;
        }
    }
}

// pack f32 weight [M,Kc] -> zero-padded f16 [Mp,Kp]
__global__ void pack_weight_kernel(const float* __restrict__ src, _Float16* __restrict__ dst,
                                   int M, int Kc, int Mp, int Kp) {
    int i = blockIdx.x * blockDim.x + threadIdx.x;
    int total = Mp * Kp;
    if (i >= total) return;
    int k = i % Kp, m = i / Kp;
    dst[i] = (m < M && k < Kc) ? (_Float16)src[(size_t)m * Kc + k] : (_Float16)0.f;
}

// ---------------------------------------------------------------------------
// attention: sim = softmax_k(g . h), affine = sim @ nbr.  64 threads per (b,n).
// g/h/F are K-packed f16.
// ---------------------------------------------------------------------------
__global__ void attention_kernel(const _Float16* __restrict__ g, const _Float16* __restrict__ h,
                                 const _Float16* __restrict__ Fh, float* __restrict__ affine,
                                 int N, int B) {
    __shared__ float sim[64];
    int bn = blockIdx.x;
    int n = bn % N, b = bn / N;
    int tid = threadIdx.x;
    int r = tid >> 4, k = tid & 15;
    size_t PG = (size_t)N * 4, PH = (size_t)N * 16;
    const _Float16* gb = g + (size_t)b * 512 * PG;
    const _Float16* hb = h + (size_t)b * 512 * PH;
    float s = 0.f;
    for (int c = 0; c < 512; ++c)
        s += (float)gb[pkidx(c, (size_t)n * 4 + r, PG)] *
             (float)hb[pkidx(c, (size_t)n * 16 + k, PH)];
    sim[tid] = s;
    __syncthreads();
    float mx = -3.4e38f;
    for (int kk = 0; kk < 16; ++kk) mx = fmaxf(mx, sim[r * 16 + kk]);
    float den = 0.f;
    for (int kk = 0; kk < 16; ++kk) den += expf(sim[r * 16 + kk] - mx);
    float w = expf(s - mx) / den;
    __syncthreads();
    sim[tid] = w;
    __syncthreads();
    if (tid < 12) {
        int d = tid >> 2, rr = tid & 3;
        const _Float16* Fb = Fh + (size_t)b * 1056 * PH;
        float a = 0.f;
        for (int kk = 0; kk < 16; ++kk)
            a += sim[rr * 16 + kk] * (float)Fb[pkidx(1020 + d, (size_t)n * 16 + kk, PH)];
        affine[(((size_t)b * 3 + d) * N + n) * 4 + rr] = a;
    }
}

// lprime = max over K of l (K-packed)
__global__ void lpool_kernel(const _Float16* __restrict__ l, float* __restrict__ lp,
                             int N, int K, int B) {
    int i = blockIdx.x * blockDim.x + threadIdx.x;
    int total = B * 512 * N;
    if (i >= total) return;
    int n = i % N, c = (i / N) % 512, b = i / (N * 512);
    size_t P = (size_t)N * K;
    const _Float16* p = l + (size_t)b * 512 * P + pkidx(c, (size_t)n * K, P);
    float m = (float)p[0];
    for (int k = 1; k < K; ++k) m = fmaxf(m, (float)p[(size_t)k * 32]);
    lp[i] = m;
}

// feat(f16, K-packed, 544 slots) = [lprime bcast (512) | affine (3) | zeros]
__global__ void feat_kernel(const float* __restrict__ lp, const float* __restrict__ affine,
                            _Float16* __restrict__ feath, int N, int R, int B) {
    const int CP = 544;
    int i = blockIdx.x * blockDim.x + threadIdx.x;
    int total = B * CP * N * R;
    if (i >= total) return;
    int rr = i % R, n = (i / R) % N, c = (i / (R * N)) % CP, b = i / (R * N * CP);
    float v = 0.f;
    if (c < 512)      v = lp[((size_t)b * 512 + c) * N + n];
    else if (c < 515) v = affine[(((size_t)b * 3 + (c - 512)) * N + n) * R + rr];
    size_t P = (size_t)N * R;
    feath[(size_t)b * CP * P + pkidx(c, (size_t)n * R + rr, P)] = (_Float16)v;
}

// r4 (3x128) + coarse: out[b, n*R+r, d] = affine + (r4w @ r3 + r4b); r3 K-packed
__global__ void final_kernel(const _Float16* __restrict__ r3, const float* __restrict__ r4w,
                             const float* __restrict__ r4b, const float* __restrict__ affine,
                             float* __restrict__ out, int N, int R, int B) {
    int i = blockIdx.x * blockDim.x + threadIdx.x;
    int total = B * N * R;
    if (i >= total) return;
    int p = i % (N * R), b = i / (N * R);
    int n = p / R, rr = p % R;
    size_t P = (size_t)N * R;
    const _Float16* rb = r3 + (size_t)b * 128 * P;
    for (int d = 0; d < 3; ++d) {
        float s = r4b[d];
        for (int c = 0; c < 128; ++c)
            s += r4w[d * 128 + c] * (float)rb[pkidx(c, (size_t)p, P)];
        out[((size_t)b * N * R + p) * 3 + d] =
            affine[(((size_t)b * 3 + d) * N + n) * R + rr] + s;
    }
}

// ---------------------------------------------------------------------------
extern "C" void kernel_launch(void* const* d_in, const int* in_sizes, int n_in,
                              void* d_out, int out_size, void* d_ws, size_t ws_size,
                              hipStream_t stream) {
    (void)in_sizes; (void)n_in; (void)out_size; (void)ws_size;
    const int B = 2, N = 1024, K = 16, R = 4, CF = 1027, FD = 339;
    const int CFP = 1056;   // 1027 padded to 32
    const int C5P = 544;    // 515 padded to 32
    const int NK = N * K, NR = N * R;
    const float* x = (const float*)d_in[0];

    auto cdiv = [](int a, int b) { return (a + b - 1) / b; };

    // ---- workspace bump allocator (256B aligned) ----
    char* ws = (char*)d_ws;
    size_t off = 0;
    auto alloc = [&](size_t bytes) -> void* {
        void* p = ws + off;
        off += (bytes + 255) & ~(size_t)255;
        return p;
    };

    int*       idx  = (int*)alloc((size_t)B * N * K * 4);
    float*     fA   = (float*)alloc((size_t)B * FD * N * 4);
    float*     fB   = (float*)alloc((size_t)B * FD * N * 4);
    float*     pre  = (float*)alloc((size_t)B * 24 * N * 4);
    float*     bmean= (float*)alloc(64 * 4);
    float*     brstd= (float*)alloc(64 * 4);
    float*     y48  = (float*)alloc((size_t)B * 48 * NK * 4);
    float*     m1o  = (float*)alloc((size_t)B * 12 * NK * 4);
    float*     m2o  = (float*)alloc((size_t)B * 12 * NK * 4);
    float*     m3o  = (float*)alloc((size_t)B * 12 * NK * 4);
    _Float16*  Fh   = (_Float16*)alloc((size_t)B * CFP * NK * 2);
    _Float16*  rh   = (_Float16*)alloc((size_t)B * CFP * NR * 2);
    _Float16*  hout = (_Float16*)alloc((size_t)B * 512 * NK * 2);
    _Float16*  lout = (_Float16*)alloc((size_t)B * 512 * NK * 2);
    _Float16*  gout = (_Float16*)alloc((size_t)B * 512 * NR * 2);
    float*     lp   = (float*)alloc((size_t)B * 512 * N * 4);
    float*     aff  = (float*)alloc((size_t)B * 3 * NR * 4);
    _Float16*  feath= (_Float16*)alloc((size_t)B * C5P * NR * 2);
    _Float16*  r1h  = (_Float16*)alloc((size_t)B * C5P * NR * 2);
    _Float16*  r2h  = (_Float16*)alloc((size_t)B * C5P * NR * 2);
    _Float16*  r3h  = (_Float16*)alloc((size_t)B * 128 * NR * 2);
    _Float16*  wgh  = (_Float16*)alloc((size_t)512 * CFP * 2);
    _Float16*  whh  = (_Float16*)alloc((size_t)512 * CFP * 2);
    _Float16*  wlh  = (_Float16*)alloc((size_t)512 * CFP * 2);
    _Float16*  w1h  = (_Float16*)alloc((size_t)640 * C5P * 2);
    _Float16*  w2h  = (_Float16*)alloc((size_t)640 * C5P * 2);
    _Float16*  w3h  = (_Float16*)alloc((size_t)128 * C5P * 2);

    // ---- parameter pointers ----
    const float* g_w  = (const float*)d_in[45];
    const float* g_b  = (const float*)d_in[46];
    const float* h_w  = (const float*)d_in[47];
    const float* h_b  = (const float*)d_in[48];
    const float* l_w  = (const float*)d_in[49];
    const float* l_b  = (const float*)d_in[50];
    const float* r1_w = (const float*)d_in[51];
    const float* r1_b = (const float*)d_in[52];
    const float* r2_w = (const float*)d_in[53];
    const float* r2_b = (const float*)d_in[54];
    const float* r3_w = (const float*)d_in[55];
    const float* r3_b = (const float*)d_in[56];
    const float* r4_w = (const float*)d_in[57];
    const float* r4_b = (const float*)d_in[58];

    auto packw = [&](const float* src, _Float16* dst, int M, int Kc, int Mp, int Kp) {
        pack_weight_kernel<<<cdiv(Mp * Kp, 256), 256, 0, stream>>>(src, dst, M, Kc, Mp, Kp);
    };
    packw(g_w,  wgh, 512, CF,  512, CFP);
    packw(h_w,  whh, 512, CF,  512, CFP);
    packw(l_w,  wlh, 512, CF,  512, CFP);
    packw(r1_w, w1h, 515, 515, 640, C5P);
    packw(r2_w, w2h, 515, 515, 640, C5P);
    packw(r3_w, w3h, 128, 515, 128, C5P);

    // ---- kNN ----
    knn_kernel<<<B * N, 256, 0, stream>>>(x, idx, N, K, B);

    // ---- 4 FEC stages ----
    const float* fin = x;
    int cin = 3;
    float* fout = fA;
    for (int s = 0; s < 4; ++s) {
        int base = 1 + 11 * s;
        const float* pw  = (const float*)d_in[base + 0];
        const float* pbi = (const float*)d_in[base + 1];
        const float* pg  = (const float*)d_in[base + 2];
        const float* pbt = (const float*)d_in[base + 3];
        const float* m1w = (const float*)d_in[base + 4];
        const float* m1g = (const float*)d_in[base + 5];
        const float* m1b = (const float*)d_in[base + 6];
        const float* m2w = (const float*)d_in[base + 7];
        const float* m2g = (const float*)d_in[base + 8];
        const float* m2b = (const float*)d_in[base + 9];
        const float* m3w = (const float*)d_in[base + 10];

        conv_pre_kernel<<<cdiv(B * 24 * N, 256), 256, 0, stream>>>(fin, pw, pbi, pre, cin, 24, N, B);
        bn_stats_kernel<<<24, 256, 0, stream>>>(pre, bmean, brstd, 24, N, B);
        bn_apply_relu_kernel<<<cdiv(B * 24 * N, 256), 256, 0, stream>>>(pre, bmean, brstd, pg, pbt, 24, N, B);
        group48_kernel<<<cdiv(B * 48 * NK, 256), 256, 0, stream>>>(pre, idx, y48, N, K, B);

        conv_nbr_kernel<<<cdiv(B * 12 * NK, 256), 256, 0, stream>>>(
            y48, 48, nullptr, 0, nullptr, 0, m1w, m1o, 12, NK, B);
        bn_stats_kernel<<<12, 256, 0, stream>>>(m1o, bmean, brstd, 12, NK, B);
        bn_apply_relu_kernel<<<cdiv(B * 12 * NK, 256), 256, 0, stream>>>(m1o, bmean, brstd, m1g, m1b, 12, NK, B);

        conv_nbr_kernel<<<cdiv(B * 12 * NK, 256), 256, 0, stream>>>(
            m1o, 12, y48, 48, nullptr, 0, m2w, m2o, 12, NK, B);
        bn_stats_kernel<<<12, 256, 0, stream>>>(m2o, bmean, brstd, 12, NK, B);
        bn_apply_relu_kernel<<<cdiv(B * 12 * NK, 256), 256, 0, stream>>>(m2o, bmean, brstd, m2g, m2b, 12, NK, B);

        conv_nbr_kernel<<<cdiv(B * 12 * NK, 256), 256, 0, stream>>>(
            m2o, 12, m1o, 12, y48, 48, m3w, m3o, 12, NK, B);

        int cout = 84 + cin;
        pool_concat_kernel<<<cdiv(B * cout * N, 256), 256, 0, stream>>>(
            m3o, m2o, m1o, y48, fin, cin, fout, N, K, B);
        fin = fout;
        cin = cout;
        fout = (fout == fA) ? fB : fA;
    }

    // ---- EVA upsampling ----
    buildF_kernel<<<cdiv(B * NK, 256), 256, 0, stream>>>(x, fin, idx, Fh, N, K, B);
    subsample_kernel<<<cdiv(B * CFP * N, 256), 256, 0, stream>>>(Fh, rh, N, K, R, B);

    auto gemm = [&](const _Float16* W, const _Float16* X, const float* bias, _Float16* O,
                    int M, int Kp, int P, int Mout, int relu) {
        dim3 grid(P / 64, cdiv(M, 128), B);
        wmma_gemm_kernel<<<grid, 256, 0, stream>>>(W, X, bias, O, M, Kp, P, Mout, relu);
    };

    gemm(whh, Fh, h_b, hout, 512, CFP, NK, 512, 1);   // h = relu(conv(F))
    gemm(wlh, Fh, l_b, lout, 512, CFP, NK, 512, 1);   // l = relu(conv(F))
    gemm(wgh, rh, g_b, gout, 512, CFP, NR, 512, 1);   // g = relu(conv(r))

    attention_kernel<<<B * N, 64, 0, stream>>>(gout, hout, Fh, aff, N, B);
    lpool_kernel<<<cdiv(B * 512 * N, 256), 256, 0, stream>>>(lout, lp, N, K, B);
    feat_kernel<<<cdiv(B * C5P * NR, 256), 256, 0, stream>>>(lp, aff, feath, N, R, B);

    // ---- coordinate reconstruction ----
    gemm(w1h, feath, r1_b, r1h, 515, C5P, NR, C5P, 1);
    gemm(w2h, r1h,   r2_b, r2h, 515, C5P, NR, C5P, 1);
    gemm(w3h, r2h,   r3_b, r3h, 128, C5P, NR, 128, 1);

    final_kernel<<<cdiv(B * NR, 256), 256, 0, stream>>>(
        r3h, r4_w, r4_b, aff, (float*)d_out, N, R, B);
}
